// _HybridBlock_44092134261078
// MI455X (gfx1250) — matrix operand order repaired and checked by reading, not compiled, not run on hardware
//
#include <hip/hip_runtime.h>

#define WG 256

typedef __attribute__((ext_vector_type(16))) __bf16 v16bf;
typedef __attribute__((ext_vector_type(8)))  float  v8f;

static __device__ __forceinline__ unsigned short f2bf(float f) {
  unsigned u = __float_as_uint(f);
  return (unsigned short)((u + 0x7FFFu + ((u >> 16) & 1u)) >> 16);
}
static __device__ __forceinline__ float bf2f(unsigned short s) {
  return __uint_as_float(((unsigned)s) << 16);
}
static __device__ __forceinline__ __bf16 bf2h(unsigned short s) {
  union { unsigned short u; __bf16 h; } c; c.u = s; return c.h;
}

// issue one per-lane 16B async copy global -> LDS (tracked by ASYNCcnt)
static __device__ __forceinline__ void async_copy_b128(const void* gptr, void* lptr) {
  unsigned      lds32 = (unsigned)(size_t)lptr;
  unsigned long long ga = (unsigned long long)(size_t)gptr;
  asm volatile("global_load_async_to_lds_b128 %0, %1, off"
               :: "v"(lds32), "v"(ga) : "memory");
}
static __device__ __forceinline__ void wait_async0() {
  asm volatile("s_wait_asynccnt 0x0" ::: "memory");
}

// ---------------- f32 -> bf16 conversion (vectorized, n % 4 == 0) ----------------
__global__ void k_cvt_bf16(const float* __restrict__ s, unsigned short* __restrict__ d, long n4) {
  for (long i = (long)blockIdx.x * blockDim.x + threadIdx.x; i < n4;
       i += (long)gridDim.x * blockDim.x) {
    float4 f = ((const float4*)s)[i];
    ushort4 o;
    o.x = f2bf(f.x); o.y = f2bf(f.y); o.z = f2bf(f.z); o.w = f2bf(f.w);
    ((ushort4*)d)[i] = o;
  }
}

// ---------------- generic bf16 WMMA GEMM ----------------
// C[m,n] = act( scale * sum_k A[m,k]*B'[k,n] + bias[n] + addScale*Add[m,n] )
// TRANSB=1: B stored [N,K]; TRANSB=0: B stored [K,N].
// Batch z decomposes as zb=z/Hdim, zh=z%Hdim with two-level strides (per-head slicing).
// 256 threads = 8 waves; 64x64 C tile; 32-deep K steps; double-buffered LDS with
// async global->LDS copies so tile s+1 streams in while tile s runs on the WMMA pipes.
template <int TRANSB>
static __device__ __forceinline__ void stage_tile(
    const unsigned short* __restrict__ A, int lda,
    const unsigned short* __restrict__ Bm, int ldb,
    unsigned short (*Als)[40], unsigned short (*Bls)[40],
    int bm, int bn, int k0, int sr, int sc, int tc, int tn, int moreK)
{
  async_copy_b128(A + (long)(bm + sr) * lda + (k0 + sc), &Als[sr][sc]);
  if (TRANSB) {
    async_copy_b128(Bm + (long)(bn + sr) * ldb + (k0 + sc), &Bls[sr][sc]);
  } else {
    const unsigned short* pb = Bm + (long)(k0 + tc) * ldb + (bn + tn);
    int4 vv = *(const int4*)pb;
    unsigned short tmp[8];
    *(int4*)tmp = vv;
#pragma unroll
    for (int j = 0; j < 8; ++j) Bls[tn + j][tc] = tmp[j];   // transpose into [n][k]
    if (moreK) __builtin_prefetch(pb + (long)32 * ldb, 0, 1);
  }
}

template <int TRANSB, int CF32>
__global__ void __launch_bounds__(256)
k_gemm(const unsigned short* __restrict__ A, long aSB, long aSH, int lda,
       const unsigned short* __restrict__ Bm, long bSB, long bSH, int ldb,
       void* __restrict__ Cv, long cSB, long cSH, int ldc,
       const float* __restrict__ bias,
       const unsigned short* __restrict__ Add, long adSB, long adSH, int ldadd, float addScale,
       float scale, int K, int Hdim, int act)
{
  // row pitch 40 halves = 80B: multiple of 16B so b128 LDS ops stay aligned.
  __shared__ unsigned short Als[2][64][40];
  __shared__ unsigned short Bls[2][64][40];

  int z = blockIdx.z, zb = z / Hdim, zh = z % Hdim;
  A  += zb * aSB + zh * aSH;
  Bm += zb * bSB + zh * bSH;
  const unsigned short* Ap = Add ? (Add + zb * adSB + zh * adSH) : nullptr;
  long cOff = zb * cSB + zh * cSH;

  int bm = blockIdx.y * 64, bn = blockIdx.x * 64;
  int tid   = threadIdx.x;
  int lane  = tid & 31;
  int wave  = tid >> 5;
  int wm    = wave >> 1;          // 4 m sub-tiles
  int wn2   = (wave & 1) * 2;     // each wave: 2 adjacent n sub-tiles
  int lcol  = lane & 15;
  int lhalf = lane >> 4;

  // per-thread staging coordinates (one 16B vector each for A and row-major B)
  int sr  = tid >> 2, sc  = (tid & 3) * 8;   // A / B(T) : 64 rows x 4 vectors
  int tc  = tid >> 3, tn  = (tid & 7) * 8;   // B(N)     : 32 k-rows x 8 vectors

  v8f acc0 = {0.f, 0.f, 0.f, 0.f, 0.f, 0.f, 0.f, 0.f};
  v8f acc1 = acc0;

  int nsteps = K >> 5;
  stage_tile<TRANSB>(A, lda, Bm, ldb, Als[0], Bls[0], bm, bn, 0, sr, sc, tc, tn, nsteps > 1);
  int cur = 0;

  for (int s = 0; s < nsteps; ++s) {
    wait_async0();         // this wave's LDS writes (tile s) are complete
    __syncthreads();       // everyone's are
    if (s + 1 < nsteps)    // stream tile s+1 into the other buffer during compute
      stage_tile<TRANSB>(A, lda, Bm, ldb, Als[cur ^ 1], Bls[cur ^ 1],
                         bm, bn, (s + 1) << 5, sr, sc, tc, tn, s + 2 < nsteps);

    // A fragment: 16x32 bf16 (lanes 0-15: K0..15; lanes 16-31: K16..31)
    v16bf af;
    int mrow = wm * 16 + lcol, kb = lhalf * 16;
#pragma unroll
    for (int j = 0; j < 8; ++j) {
      af[j]     = bf2h(Als[cur][mrow][kb + j]);
      af[8 + j] = bf2h(Als[cur][mrow][kb + 8 + j]);
    }
    // B fragments: 32x16, lanes 0-15 hold K0..15, lanes 16-31 hold K16..31
    v16bf bf0, bf1;
    int n0 = wn2 * 16 + lcol, n1 = n0 + 16, kb2 = lhalf * 16;
#pragma unroll
    for (int e = 0; e < 16; ++e) {
      bf0[e] = bf2h(Bls[cur][n0][kb2 + e]);
      bf1[e] = bf2h(Bls[cur][n1][kb2 + e]);
    }
    acc0 = __builtin_amdgcn_wmma_f32_16x16x32_bf16(false, af, false, bf0, (short)0, acc0, false, false);
    acc1 = __builtin_amdgcn_wmma_f32_16x16x32_bf16(false, af, false, bf1, (short)0, acc1, false, false);
    cur ^= 1;
  }

  float* Cf          = (float*)Cv + cOff;
  unsigned short* Ch = (unsigned short*)Cv + cOff;
#pragma unroll
  for (int t2 = 0; t2 < 2; ++t2) {
    int nn = bn + (wn2 + t2) * 16 + lcol;
#pragma unroll
    for (int r = 0; r < 8; ++r) {
      int mm = bm + wm * 16 + lhalf * 8 + r;   // C layout: VGPR r -> M=r / r+8
      float v = ((t2 == 0) ? acc0[r] : acc1[r]) * scale;
      if (bias) v += bias[nn];
      if (Ap)   v += addScale * bf2f(Ap[(long)mm * ldadd + nn]);
      if (act == 1)      v = tanhf(v);
      else if (act == 2) v = 1.f / (1.f + __expf(-v));
      if (CF32) Cf[(long)mm * ldc + nn] = v;
      else      Ch[(long)mm * ldc + nn] = f2bf(v);
    }
  }
}

// ---------------- row softmax (f32 in -> bf16 probabilities out) ----------------
__global__ void k_softmax_bf16(const float* __restrict__ S, unsigned short* __restrict__ P, int len) {
  __shared__ float red[WG];
  long row = blockIdx.x;
  const float* s = S + row * (long)len;
  unsigned short* p = P + row * (long)len;
  int t = threadIdx.x;
  float m = -3.4e38f;
  for (int i = t; i < len; i += WG) m = fmaxf(m, s[i]);
  red[t] = m; __syncthreads();
  for (int o = WG / 2; o; o >>= 1) { if (t < o) red[t] = fmaxf(red[t], red[t + o]); __syncthreads(); }
  m = red[0]; __syncthreads();
  float sum = 0.f;
  for (int i = t; i < len; i += WG) sum += __expf(s[i] - m);
  red[t] = sum; __syncthreads();
  for (int o = WG / 2; o; o >>= 1) { if (t < o) red[t] += red[t + o]; __syncthreads(); }
  float inv = 1.f / red[0];
  for (int i = t; i < len; i += WG) p[i] = f2bf(__expf(s[i] - m) * inv);
}

// ---------------- exact top-K mask + row-normalized adjacency ----------------
__global__ void k_topk_rownorm(const float* __restrict__ Aadj, unsigned short* __restrict__ P,
                               int NI, int KTOP) {
  __shared__ float vals[256], work[256], msk[256], redv[256];
  __shared__ int redi[256];
  int v = blockIdx.x, t = threadIdx.x;
  vals[t] = Aadj[(long)v * NI + t]; work[t] = vals[t]; msk[t] = 0.f;
  __syncthreads();
  for (int k = 0; k < KTOP; ++k) {
    redv[t] = work[t]; redi[t] = t; __syncthreads();
    for (int o = 128; o; o >>= 1) {
      if (t < o && redv[t + o] > redv[t]) { redv[t] = redv[t + o]; redi[t] = redi[t + o]; }
      __syncthreads();
    }
    if (t == 0) { msk[redi[0]] = 1.f; work[redi[0]] = -3.4e38f; }
    __syncthreads();
  }
  float adj = vals[t] * msk[t] + (t == v ? 1.f : 0.f);
  redv[t] = adj; __syncthreads();
  for (int o = 128; o; o >>= 1) { if (t < o) redv[t] += redv[t + o]; __syncthreads(); }
  P[(long)v * NI + t] = f2bf(adj / redv[0]);
}

// ---------------- residual + LayerNorm over D=512 ----------------
__global__ void k_residual_ln(const float* __restrict__ x, const float* __restrict__ y,
                              const float* __restrict__ g, const float* __restrict__ b,
                              float* __restrict__ out, int D) {
  __shared__ float red[WG];
  long row = blockIdx.x;
  const float* xr = x + row * D;
  const float* yr = y + row * D;
  float* o = out + row * D;
  int t = threadIdx.x;
  float v0 = xr[t] + yr[t];
  float v1 = xr[t + 256] + yr[t + 256];
  red[t] = v0 + v1; __syncthreads();
  for (int of = 128; of; of >>= 1) { if (t < of) red[t] += red[t + of]; __syncthreads(); }
  float mu = red[0] / (float)D; __syncthreads();
  float d0 = v0 - mu, d1 = v1 - mu;
  red[t] = d0 * d0 + d1 * d1; __syncthreads();
  for (int of = 128; of; of >>= 1) { if (t < of) red[t] += red[t + of]; __syncthreads(); }
  float inv = rsqrtf(red[0] / (float)D + 1e-5f);
  o[t]       = d0 * inv * g[t]       + b[t];
  o[t + 256] = d1 * inv * g[t + 256] + b[t + 256];
}

// ====================== host orchestration ======================
extern "C" void kernel_launch(void* const* d_in, const int* in_sizes, int n_in,
                              void* d_out, int out_size, void* d_ws, size_t ws_size,
                              hipStream_t stream) {
  (void)in_sizes; (void)n_in; (void)out_size; (void)ws_size;
  constexpr int Bc = 16, Nn = 2048, Dd = 512, Hh = 8, DHd = 64, Ff = 64, NI = 256;
  constexpr int KTOP = 179;
  constexpr long ND  = (long)Nn * Dd;       // 1,048,576
  constexpr long BND = (long)Bc * ND;       // 16,777,216
  constexpr long FD  = (long)Ff * Dd;       // 32,768

  const float* x      = (const float*)d_in[0];
  const float* route  = (const float*)d_in[1];
  const float* down_w = (const float*)d_in[2];  const float* down_b = (const float*)d_in[3];
  const float* up_w   = (const float*)d_in[4];  const float* up_b   = (const float*)d_in[5];
  const float* sq_w   = (const float*)d_in[6];  const float* sq_b   = (const float*)d_in[7];
  const float* sk_w   = (const float*)d_in[8];  const float* sk_b   = (const float*)d_in[9];
  const float* sv_w   = (const float*)d_in[10]; const float* sv_b   = (const float*)d_in[11];
  const float* sp_w   = (const float*)d_in[12]; const float* sp_b   = (const float*)d_in[13];
  const float* rq_w   = (const float*)d_in[14]; const float* rq_b   = (const float*)d_in[15];
  const float* rk_w   = (const float*)d_in[16]; const float* rk_b   = (const float*)d_in[17];
  const float* rv_w   = (const float*)d_in[18]; const float* rv_b   = (const float*)d_in[19];
  const float* rp_w   = (const float*)d_in[20]; const float* rp_b   = (const float*)d_in[21];
  const float* l1_w   = (const float*)d_in[22]; const float* l1_b   = (const float*)d_in[23];
  const float* l2_w   = (const float*)d_in[24]; const float* l2_b   = (const float*)d_in[25];
  const float* head_w = (const float*)d_in[26]; const float* head_b = (const float*)d_in[27];
  const float* e1     = (const float*)d_in[28]; const float* e2     = (const float*)d_in[29];
  const float* ln_g   = (const float*)d_in[30]; const float* ln_b   = (const float*)d_in[31];

  size_t off = 0;
  auto alloc = [&](size_t bytes) -> void* {
    size_t o = (off + 255) & ~(size_t)255;
    off = o + bytes;
    return (void*)((char*)d_ws + o);
  };
  typedef unsigned short us;
  us* xb     = (us*)alloc(BND * 2);
  us* routeb = (us*)alloc(FD * 2);
  us* downb  = (us*)alloc((size_t)NI * Dd * 2);
  us* upb    = (us*)alloc((size_t)Dd * NI * 2);
  us* sqb    = (us*)alloc((size_t)Dd * Dd * 2);
  us* skb    = (us*)alloc((size_t)Dd * Dd * 2);
  us* svb    = (us*)alloc((size_t)Dd * Dd * 2);
  us* spb    = (us*)alloc((size_t)Dd * Dd * 2);
  us* rqb    = (us*)alloc((size_t)Dd * Dd * 2);
  us* rkb    = (us*)alloc((size_t)Dd * Dd * 2);
  us* rvb    = (us*)alloc((size_t)Dd * Dd * 2);
  us* rpb    = (us*)alloc((size_t)Dd * Dd * 2);
  us* l1b    = (us*)alloc((size_t)NI * NI * 2);
  us* l2b    = (us*)alloc((size_t)NI * NI * 2);
  us* headb  = (us*)alloc((size_t)NI * 3 * NI * 2);
  us* e1b    = (us*)alloc((size_t)NI * NI * 2);
  us* e2b    = (us*)alloc((size_t)NI * NI * 2);
  us* Qr     = (us*)alloc(FD * 2);
  us* Kx     = (us*)alloc(BND * 2);
  us* Vx     = (us*)alloc(BND * 2);
  float* S1  = (float*)alloc((size_t)Bc * Hh * Ff * Nn * 4);
  us* P1     = (us*)alloc((size_t)Bc * Hh * Ff * Nn * 2);
  us* O1     = (us*)alloc((size_t)Bc * FD * 2);
  us* bufb   = (us*)alloc((size_t)Bc * FD * 2);
  us* zF     = (us*)alloc((size_t)Bc * Ff * 3 * NI * 2);
  us* gF     = (us*)alloc((size_t)Bc * Ff * NI * 2);
  us* buf2   = (us*)alloc((size_t)Bc * FD * 2);
  us* v1b    = (us*)alloc((size_t)NI * NI * 2);
  us* v2b    = (us*)alloc((size_t)NI * NI * 2);
  float* Aadj = (float*)alloc((size_t)NI * NI * 4);
  us* Pmat   = (us*)alloc((size_t)NI * NI * 2);
  us* Qx     = (us*)alloc(BND * 2);
  us* Kb     = (us*)alloc((size_t)Bc * FD * 2);
  us* Vb     = (us*)alloc((size_t)Bc * FD * 2);
  float* S2  = (float*)alloc((size_t)Bc * Hh * Nn * Ff * 4);
  us* P2     = (us*)alloc((size_t)Bc * Hh * Nn * Ff * 2);
  us* O2     = (us*)alloc(BND * 2);
  float* yv  = (float*)alloc(BND * 4);

  auto cvt = [&](const float* s, us* d, long n) {
    k_cvt_bf16<<<1024, WG, 0, stream>>>(s, d, n / 4);
  };
  cvt(x, xb, BND);
  cvt(route, routeb, FD);
  cvt(down_w, downb, (long)NI * Dd);
  cvt(up_w, upb, (long)Dd * NI);
  cvt(sq_w, sqb, (long)Dd * Dd);  cvt(sk_w, skb, (long)Dd * Dd);
  cvt(sv_w, svb, (long)Dd * Dd);  cvt(sp_w, spb, (long)Dd * Dd);
  cvt(rq_w, rqb, (long)Dd * Dd);  cvt(rk_w, rkb, (long)Dd * Dd);
  cvt(rv_w, rvb, (long)Dd * Dd);  cvt(rp_w, rpb, (long)Dd * Dd);
  cvt(l1_w, l1b, (long)NI * NI);  cvt(l2_w, l2b, (long)NI * NI);
  cvt(head_w, headb, (long)NI * 3 * NI);
  cvt(e1, e1b, (long)NI * NI);    cvt(e2, e2b, (long)NI * NI);

  auto gemm = [&](const us* A, long aSB, long aSH, int lda,
                  const us* Bp, long bSB, long bSH, int ldb, int tB,
                  void* C, long cSB, long cSH, int ldc, int cF32,
                  const float* bias,
                  const us* Add, long adSB, long adSH, int ldadd, float addScale,
                  float scale, int M, int Nc, int K, int batch, int Hdim, int act) {
    dim3 g(Nc / 64, M / 64, batch);
    if (tB) {
      if (cF32) k_gemm<1, 1><<<g, WG, 0, stream>>>(A, aSB, aSH, lda, Bp, bSB, bSH, ldb,
                    C, cSB, cSH, ldc, bias, Add, adSB, adSH, ldadd, addScale, scale, K, Hdim, act);
      else      k_gemm<1, 0><<<g, WG, 0, stream>>>(A, aSB, aSH, lda, Bp, bSB, bSH, ldb,
                    C, cSB, cSH, ldc, bias, Add, adSB, adSH, ldadd, addScale, scale, K, Hdim, act);
    } else {
      if (cF32) k_gemm<0, 1><<<g, WG, 0, stream>>>(A, aSB, aSH, lda, Bp, bSB, bSH, ldb,
                    C, cSB, cSH, ldc, bias, Add, adSB, adSH, ldadd, addScale, scale, K, Hdim, act);
      else      k_gemm<0, 0><<<g, WG, 0, stream>>>(A, aSB, aSH, lda, Bp, bSB, bSH, ldb,
                    C, cSB, cSH, ldc, bias, Add, adSB, adSH, ldadd, addScale, scale, K, Hdim, act);
    }
  };

  const float sc = 0.125f;              // DH^-0.5
  // ---- send attention ----
  gemm(routeb,0,0,Dd, sqb,0,0,Dd,1, Qr,0,0,Dd,0, sq_b, nullptr,0,0,0,0.f, 1.f, Ff,Dd,Dd, 1,1,0);
  gemm(xb,ND,0,Dd,    skb,0,0,Dd,1, Kx,ND,0,Dd,0, sk_b, nullptr,0,0,0,0.f, 1.f, Nn,Dd,Dd, Bc,1,0);
  gemm(xb,ND,0,Dd,    svb,0,0,Dd,1, Vx,ND,0,Dd,0, sv_b, nullptr,0,0,0,0.f, 1.f, Nn,Dd,Dd, Bc,1,0);
  gemm(Qr,0,DHd,Dd, Kx,ND,DHd,Dd,1, S1,(long)Hh*Ff*Nn,(long)Ff*Nn,Nn,1, nullptr,
       nullptr,0,0,0,0.f, sc, Ff,Nn,DHd, Bc*Hh,Hh,0);
  k_softmax_bf16<<<Bc * Hh * Ff, WG, 0, stream>>>(S1, P1, Nn);
  gemm(P1,(long)Hh*Ff*Nn,(long)Ff*Nn,Nn, Vx,ND,DHd,Dd,0, O1,FD,DHd,Dd,0, nullptr,
       nullptr,0,0,0,0.f, 1.f, Ff,DHd,Nn, Bc*Hh,Hh,0);
  gemm(O1,FD,0,Dd, spb,0,0,Dd,1, bufb,FD,0,Dd,0, sp_b, nullptr,0,0,0,0.f, 1.f, Ff,Dd,Dd, Bc,1,0);

  // ---- graph branch, all in F-major layout ([B, F, node]) so no transposes ----
  long zstr = (long)Ff * 3 * NI;
  gemm(bufb,FD,0,Dd, downb,0,0,Dd,1, zF,zstr,0,3*NI,0, down_b, nullptr,0,0,0,0.f,
       1.f, Ff,NI,Dd, Bc,1,0);                                           // hF -> zF[:, :, 0:NI]
  gemm(e1b,0,0,NI, l1b,0,0,NI,1, v1b,0,0,NI,0, l1_b, nullptr,0,0,0,0.f, 1.f, NI,NI,NI, 1,1,1);
  gemm(e2b,0,0,NI, l2b,0,0,NI,1, v2b,0,0,NI,0, l2_b, nullptr,0,0,0,0.f, 1.f, NI,NI,NI, 1,1,1);
  gemm(v1b,0,0,NI, v2b,0,0,NI,1, Aadj,0,0,NI,1, nullptr, nullptr,0,0,0,0.f, 1.f, NI,NI,NI, 1,1,2);
  k_topk_rownorm<<<NI, NI, 0, stream>>>(Aadj, Pmat, NI, KTOP);
  // MixProp: base_{t+1} = 0.05*h + 0.95*(base_t @ P^T)
  gemm(zF,zstr,0,3*NI,      Pmat,0,0,NI,1, zF+NI,  zstr,0,3*NI,0, nullptr,
       zF,zstr,0,3*NI,0.05f, 0.95f, Ff,NI,NI, Bc,1,0);
  gemm(zF+NI,zstr,0,3*NI,   Pmat,0,0,NI,1, zF+2*NI,zstr,0,3*NI,0, nullptr,
       zF,zstr,0,3*NI,0.05f, 0.95f, Ff,NI,NI, Bc,1,0);
  gemm(zF,zstr,0,3*NI, headb,0,0,3*NI,1, gF,(long)Ff*NI,0,NI,0, head_b,
       nullptr,0,0,0,0.f, 1.f, Ff,NI,3*NI, Bc,1,0);
  gemm(gF,(long)Ff*NI,0,NI, upb,0,0,NI,1, buf2,FD,0,Dd,0, up_b,
       nullptr,0,0,0,0.f, 1.f, Ff,Dd,NI, Bc,1,0);

  // ---- recv attention ----
  gemm(xb,ND,0,Dd,   rqb,0,0,Dd,1, Qx,ND,0,Dd,0, rq_b, nullptr,0,0,0,0.f, 1.f, Nn,Dd,Dd, Bc,1,0);
  gemm(buf2,FD,0,Dd, rkb,0,0,Dd,1, Kb,FD,0,Dd,0, rk_b, nullptr,0,0,0,0.f, 1.f, Ff,Dd,Dd, Bc,1,0);
  gemm(buf2,FD,0,Dd, rvb,0,0,Dd,1, Vb,FD,0,Dd,0, rv_b, nullptr,0,0,0,0.f, 1.f, Ff,Dd,Dd, Bc,1,0);
  gemm(Qx,ND,DHd,Dd, Kb,FD,DHd,Dd,1, S2,(long)Hh*Nn*Ff,(long)Nn*Ff,Ff,1, nullptr,
       nullptr,0,0,0,0.f, sc, Nn,Ff,DHd, Bc*Hh,Hh,0);
  k_softmax_bf16<<<Bc * Hh * Nn, WG, 0, stream>>>(S2, P2, Ff);
  gemm(P2,(long)Hh*Nn*Ff,(long)Nn*Ff,Ff, Vb,FD,DHd,Dd,0, O2,ND,DHd,Dd,0, nullptr,
       nullptr,0,0,0,0.f, 1.f, Nn,DHd,Ff, Bc*Hh,Hh,0);
  gemm(O2,ND,0,Dd, rpb,0,0,Dd,1, yv,ND,0,Dd,1, rp_b, nullptr,0,0,0,0.f, 1.f, Nn,Dd,Dd, Bc,1,0);

  // ---- residual + LayerNorm ----
  k_residual_ln<<<Bc * Nn, WG, 0, stream>>>(x, yv, ln_g, ln_b, (float*)d_out, Dd);
}
